// Sub1Linear_2534030705117
// MI455X (gfx1250) — matrix operand order repaired and checked by reading, not compiled, not run on hardware
//
#include <hip/hip_runtime.h>

typedef __bf16 bf16;
typedef __attribute__((ext_vector_type(16))) __bf16 v16bf;
typedef __attribute__((ext_vector_type(8)))  __bf16 v8bf;
typedef __attribute__((ext_vector_type(8)))  float  v8f;
typedef __attribute__((ext_vector_type(16))) unsigned short v16us;
typedef __attribute__((ext_vector_type(8)))  unsigned short v8us;

#define GAS __attribute__((address_space(1)))
typedef const bf16  GAS* gbf;    // global-address-space bf16 pointer
typedef const v8bf  GAS* gv8;
typedef const v16bf GAS* gv16;

#define BM 128
#define BN 128
#define TM 4   // 16-row WMMA tiles per wave in M
#define TN 2   // 16-col WMMA tiles per wave in N

__device__ __forceinline__ unsigned short tern_sel(int c, unsigned short mn,
                                                   unsigned short mx) {
    // code: 0 -> 0.0bf16 (0x0000), 1 -> min bits, 2 -> max bits
    unsigned short v = (c == 1) ? mn : (unsigned short)0;
    return (c == 2) ? mx : v;
}

// ---------------------------------------------------------------------------
// Pass 1: decode int32 ternary codes -> bf16 weight matrix in workspace.
// ---------------------------------------------------------------------------
__global__ __launch_bounds__(256) void tern_decode(
    const int* __restrict__ Ct, const unsigned short* __restrict__ MM,
    unsigned short* __restrict__ Wd, unsigned int Wcols) {
    unsigned int idx = (blockIdx.x * blockDim.x + threadIdx.x) * 8u;
    unsigned int row = idx / Wcols;
    unsigned short mn = MM[2u * row];
    unsigned short mx = MM[2u * row + 1u];

    const int4* cp = (const int4*)(Ct + idx);
    int4 q0 = cp[0];
    int4 q1 = cp[1];
    int c[8] = {q0.x, q0.y, q0.z, q0.w, q1.x, q1.y, q1.z, q1.w};

    v8us o;
#pragma unroll
    for (int i = 0; i < 8; i++) o[i] = tern_sel(c[i], mn, mx);
    *(v8us*)(Wd + idx) = o;
}

// ---------------------------------------------------------------------------
// Fragment loads with compile-time element offset -> immediate-offset b128s.
// A fragment (16x32): lane<16 holds M=lane, K = k..k+15 (contiguous);
//                     lane>=16 holds M=lane-16, K = k+16..k+31
// B fragment (32x16): lane<16 holds N=lane, K = k..k+15 (contiguous 32B);
//                     lane>=16 holds N=lane-16, K = k+16..k+31
// ---------------------------------------------------------------------------
template <int KOFF>
__device__ __forceinline__ void load_frags(const gbf* ap, const gbf* bp,
                                           v16bf* a, v16bf* b) {
#pragma unroll
    for (int t = 0; t < TM; t++) {
        v8bf lo = *(gv8)(ap[t] + KOFF);
        v8bf hi = *(gv8)(ap[t] + KOFF + 8);
#pragma unroll
        for (int i = 0; i < 8; i++) { a[t][i] = lo[i]; a[t][8 + i] = hi[i]; }
    }
#pragma unroll
    for (int u = 0; u < TN; u++) b[u] = *(gv16)(bp[u] + KOFF);
}

__device__ __forceinline__ void do_wmma(const v16bf* a, const v16bf* b,
                                        v8f acc[TM][TN]) {
#pragma unroll
    for (int t = 0; t < TM; t++)
#pragma unroll
        for (int u = 0; u < TN; u++)
            acc[t][u] = __builtin_amdgcn_wmma_f32_16x16x32_bf16(
                false, a[t], false, b[u], (short)0, acc[t][u], false, false);
}

// ---------------------------------------------------------------------------
// Pass 2: bf16 WMMA GEMM.  Y[m,n] = sum_k X[m,k] * Wd[n,k]
// Global-AS pointer induction + register ping-pong double buffering.
// ---------------------------------------------------------------------------
__global__ __launch_bounds__(256) void tern_gemm_bf16(
    const bf16* __restrict__ X, const bf16* __restrict__ Wd,
    bf16* __restrict__ Y, int K, int N) {
    const int lane = threadIdx.x & 31;
    const int wid  = threadIdx.x >> 5;   // 0..7
    const int wm   = wid & 1;            // 2 waves along M
    const int wn   = wid >> 1;           // 4 waves along N
    const int l15  = lane & 15;
    const int lh   = lane >> 4;

    const int m0 = blockIdx.x * BM + wm * (TM * 16);
    const int n0 = blockIdx.y * BN + wn * (TN * 16);

    gbf ap[TM];
#pragma unroll
    for (int t = 0; t < TM; t++)
        ap[t] = (gbf)X + (size_t)(m0 + t * 16 + l15) * K + lh * 16;
    gbf bp[TN];
#pragma unroll
    for (int u = 0; u < TN; u++)
        bp[u] = (gbf)Wd + (size_t)(n0 + u * 16 + l15) * K + lh * 16;

    v8f acc[TM][TN];
#pragma unroll
    for (int t = 0; t < TM; t++)
#pragma unroll
        for (int u = 0; u < TN; u++)
#pragma unroll
            for (int i = 0; i < 8; i++) acc[t][u][i] = 0.0f;

    v16bf a0[TM], b0[TN], a1[TM], b1[TN];
    load_frags<0>(ap, bp, a0, b0);   // prologue: K-step 0

    const int iters = K / 64 - 1;    // steady-state 64-wide K macro-steps
#pragma unroll 1
    for (int it = 0; it < iters; ++it) {
        // prefetch all six per-lane row streams 1KB ahead
        __builtin_prefetch((const bf16*)ap[0] + 512, 0, 3);
        __builtin_prefetch((const bf16*)ap[1] + 512, 0, 3);
        __builtin_prefetch((const bf16*)ap[2] + 512, 0, 3);
        __builtin_prefetch((const bf16*)ap[3] + 512, 0, 3);
        __builtin_prefetch((const bf16*)bp[0] + 512, 0, 3);
        __builtin_prefetch((const bf16*)bp[1] + 512, 0, 3);

        load_frags<32>(ap, bp, a1, b1);   // in flight during...
        do_wmma(a0, b0, acc);             // ...this WMMA batch
        load_frags<64>(ap, bp, a0, b0);   // in flight during...
        do_wmma(a1, b1, acc);             // ...this WMMA batch

#pragma unroll
        for (int t = 0; t < TM; t++) ap[t] += 64;
#pragma unroll
        for (int u = 0; u < TN; u++) bp[u] += 64;
    }

    // epilogue: final 64-wide step (buf0 already loaded, no OOB loads)
    load_frags<32>(ap, bp, a1, b1);
    do_wmma(a0, b0, acc);
    do_wmma(a1, b1, acc);

    // D layout: lane<16 -> N=lane, M=r; lane>=16 -> N=lane-16, M=r+8.
#pragma unroll
    for (int t = 0; t < TM; t++)
#pragma unroll
        for (int u = 0; u < TN; u++) {
            const int n = n0 + u * 16 + l15;
#pragma unroll
            for (int r = 0; r < 8; r++) {
                const int m = m0 + t * 16 + lh * 8 + r;
                Y[(size_t)m * N + n] = (bf16)acc[t][u][r];
            }
        }
}

// ---------------------------------------------------------------------------
// Fallback: fused decode-in-the-GEMM (used only if workspace is too small).
// ---------------------------------------------------------------------------
__global__ __launch_bounds__(256) void tern_gemm_fused(
    const bf16* __restrict__ X, const int* __restrict__ Ct,
    const unsigned short* __restrict__ MM, bf16* __restrict__ Y,
    int K, int N) {
    const int lane = threadIdx.x & 31;
    const int wid  = threadIdx.x >> 5;
    const int wm   = wid & 1;
    const int wn   = wid >> 1;
    const int l15  = lane & 15;
    const int lh   = lane >> 4;

    const int m0 = blockIdx.x * BM + wm * (TM * 16);
    const int n0 = blockIdx.y * BN + wn * (TN * 16);

    const bf16* ap[TM];
#pragma unroll
    for (int t = 0; t < TM; t++)
        ap[t] = X + (size_t)(m0 + t * 16 + l15) * K + lh * 16;

    const int* cp0[TN];
    unsigned short mn[TN], mx[TN];
#pragma unroll
    for (int u = 0; u < TN; u++) {
        const int n = n0 + u * 16 + l15;
        cp0[u] = Ct + (size_t)n * K + lh * 16;
        mn[u] = MM[2 * n];
        mx[u] = MM[2 * n + 1];
    }

    v8f acc[TM][TN];
#pragma unroll
    for (int t = 0; t < TM; t++)
#pragma unroll
        for (int u = 0; u < TN; u++)
#pragma unroll
            for (int i = 0; i < 8; i++) acc[t][u][i] = 0.0f;

#pragma unroll 1
    for (int k = 0; k < K; k += 32) {
        v16bf a[TM], b[TN];
#pragma unroll
        for (int t = 0; t < TM; t++) {
            v8bf lo = *(const v8bf*)(ap[t] + k);
            v8bf hi = *(const v8bf*)(ap[t] + k + 8);
#pragma unroll
            for (int i = 0; i < 8; i++) { a[t][i] = lo[i]; a[t][8 + i] = hi[i]; }
        }
#pragma unroll
        for (int u = 0; u < TN; u++) {
            const int4* cp = (const int4*)(cp0[u] + k);
            int4 q0 = cp[0], q1 = cp[1], q2 = cp[2], q3 = cp[3];
            int c[16] = {q0.x, q0.y, q0.z, q0.w, q1.x, q1.y, q1.z, q1.w,
                         q2.x, q2.y, q2.z, q2.w, q3.x, q3.y, q3.z, q3.w};
            v16us bu;
#pragma unroll
            for (int i = 0; i < 16; i++) bu[i] = tern_sel(c[i], mn[u], mx[u]);
            b[u] = __builtin_bit_cast(v16bf, bu);
        }
#pragma unroll
        for (int t = 0; t < TM; t++)
#pragma unroll
            for (int u = 0; u < TN; u++)
                acc[t][u] = __builtin_amdgcn_wmma_f32_16x16x32_bf16(
                    false, a[t], false, b[u], (short)0, acc[t][u], false, false);
    }

#pragma unroll
    for (int t = 0; t < TM; t++)
#pragma unroll
        for (int u = 0; u < TN; u++) {
            const int n = n0 + u * 16 + l15;
#pragma unroll
            for (int r = 0; r < 8; r++) {
                const int m = m0 + t * 16 + lh * 8 + r;
                Y[(size_t)m * N + n] = (bf16)acc[t][u][r];
            }
        }
}

extern "C" void kernel_launch(void* const* d_in, const int* in_sizes, int n_in,
                              void* d_out, int out_size, void* d_ws, size_t ws_size,
                              hipStream_t stream) {
    (void)n_in; (void)out_size;
    const bf16* x = (const bf16*)d_in[0];
    const int* codes = (const int*)d_in[1];
    const unsigned short* mm = (const unsigned short*)d_in[2];
    bf16* y = (bf16*)d_out;

    const long long H = (long long)in_sizes[2] / 2;        // 8192
    const long long W = (long long)in_sizes[1] / H;        // 8192 (= K)
    const long long B = (long long)in_sizes[0] / W;        // 2048 (= M)

    const int K = (int)W;
    const int N = (int)H;
    dim3 grid((unsigned)(B / BM), (unsigned)(H / BN));     // batch tiles fastest
    dim3 block(256);

    const size_t need = (size_t)H * (size_t)W * sizeof(unsigned short);
    if (ws_size >= need) {
        unsigned int nelem = (unsigned int)(H * W);
        unsigned int dblocks = nelem / (256u * 8u);
        tern_decode<<<dblocks, 256, 0, stream>>>(codes, mm,
                                                 (unsigned short*)d_ws,
                                                 (unsigned int)W);
        tern_gemm_bf16<<<grid, block, 0, stream>>>(x, (const bf16*)d_ws, y, K, N);
    } else {
        tern_gemm_fused<<<grid, block, 0, stream>>>(x, codes, mm, y, K, N);
    }
}
